// SequenceCompressorAttention_16192026706122
// MI455X (gfx1250) — matrix operand order repaired and checked by reading, not compiled, not run on hardware
//
#include <hip/hip_runtime.h>
#include <hip/hip_bf16.h>

// ---------------------------------------------------------------------------
// MI455X (gfx1250) fused perceiver cross-attention, bf16 WMMA everywhere.
//  x:(8,8192,1024) latents:(8,256,1024) -> out:(8,256,1024) fp32
//  Pipeline: LN -> [q gemm][kv gemm] -> flash attention -> out gemm
//  GEMMs: double-buffered LDS fed by global_load_async_to_lds_b128 (ASYNCcnt).
//  Attention: flash, row-sum folded into WMMA via a ones-tile.
// ---------------------------------------------------------------------------

typedef __bf16 bf16_t;
typedef __attribute__((ext_vector_type(16))) __bf16 bf16x16;
typedef __attribute__((ext_vector_type(8)))  __bf16 bf16x8;
typedef __attribute__((ext_vector_type(4)))  __bf16 bf16x4;
typedef __attribute__((ext_vector_type(8)))  float  f32x8;

#define DIMN   1024
#define HEADS  16
#define DHEAD  64
#define N1     8192
#define N2     256
#define NTOT   8448
#define BATCH  8

// ---- WMMA helpers ----------------------------------------------------------
__device__ __forceinline__ f32x8 wmma_bf16(bf16x16 a, bf16x16 b, f32x8 c) {
  return __builtin_amdgcn_wmma_f32_16x16x32_bf16(false, a, false, b,
                                                 (short)0, c, false, false);
}

// A fragment (16xK=32 bf16): lane holds row m=lane&15, K chunks at kk and kk+16,
// kk = (lane>>4)*8  (per ISA 16-bit A layout).
__device__ __forceinline__ bf16x16 frag_a(const bf16_t* p, int stride, int lane) {
  const int m  = lane & 15;
  const int kk = (lane >> 4) << 3;
  bf16x8 lo = *(const bf16x8*)(p + m * stride + kk);
  bf16x8 hi = *(const bf16x8*)(p + m * stride + kk + 16);
  return __builtin_shufflevector(lo, hi, 0,1,2,3,4,5,6,7,8,9,10,11,12,13,14,15);
}

// B fragment (K=32 x 16 bf16) read from an LDS tile stored [n][k]:
// lane holds col n=lane&15, K chunks at kk and kk+16, kk = (lane>>4)*8 (matches frag_a).
__device__ __forceinline__ bf16x16 frag_b(const bf16_t* p, int stride, int lane) {
  const int n  = lane & 15;
  const int kk = (lane >> 4) << 3;
  bf16x8 lo = *(const bf16x8*)(p + n * stride + kk);
  bf16x8 hi = *(const bf16x8*)(p + n * stride + kk + 16);
  return __builtin_shufflevector(lo, hi, 0,1,2,3,4,5,6,7,8,9,10,11,12,13,14,15);
}

// Async DMA: global (16B per lane) -> LDS, tracked with ASYNCcnt.
__device__ __forceinline__ void async_b128(unsigned lds_off, const bf16_t* gptr) {
  asm volatile("global_load_async_to_lds_b128 %0, %1, off"
               :: "v"(lds_off), "v"(gptr) : "memory");
}

// ---- 1) fused LayerNorm -> bf16 -------------------------------------------
__global__ void __launch_bounds__(256)
ln_kernel(const float* __restrict__ x, const float* __restrict__ lat,
          const float* __restrict__ gm, const float* __restrict__ bm,
          const float* __restrict__ gl, const float* __restrict__ bl,
          bf16_t* __restrict__ kvin, bf16_t* __restrict__ lnbuf) {
  const int row = blockIdx.x;                // 0 .. BATCH*NTOT-1
  const int b  = row / NTOT;
  const int rr = row - b * NTOT;
  const bool is_lat = rr >= N1;
  const float* src = is_lat ? lat + (size_t)(b * N2 + (rr - N1)) * DIMN
                            : x   + (size_t)(b * N1 + rr) * DIMN;
  const float* g  = is_lat ? gl : gm;
  const float* be = is_lat ? bl : bm;
  const int t = threadIdx.x;                 // 256 threads, 4 elems each
  float4 f = ((const float4*)src)[t];
  float s  = f.x + f.y + f.z + f.w;
  float s2 = f.x*f.x + f.y*f.y + f.z*f.z + f.w*f.w;
  #pragma unroll
  for (int m = 16; m >= 1; m >>= 1) { s += __shfl_xor(s, m); s2 += __shfl_xor(s2, m); }
  __shared__ float red[16];
  if ((t & 31) == 0) { red[t >> 5] = s; red[8 + (t >> 5)] = s2; }
  __syncthreads();
  float ts = 0.f, ts2 = 0.f;
  #pragma unroll
  for (int i = 0; i < 8; ++i) { ts += red[i]; ts2 += red[8 + i]; }
  const float mu  = ts * (1.f / DIMN);
  const float inv = rsqrtf(ts2 * (1.f / DIMN) - mu * mu + 1e-5f);
  float4 gv = ((const float4*)g)[t];
  float4 bv = ((const float4*)be)[t];
  bf16x4 o;
  o[0] = (bf16_t)((f.x - mu) * inv * gv.x + bv.x);
  o[1] = (bf16_t)((f.y - mu) * inv * gv.y + bv.y);
  o[2] = (bf16_t)((f.z - mu) * inv * gv.z + bv.z);
  o[3] = (bf16_t)((f.w - mu) * inv * gv.w + bv.w);
  *(bf16x4*)(kvin + (size_t)row * DIMN + t * 4) = o;
  if (is_lat)
    *(bf16x4*)(lnbuf + (size_t)(b * N2 + rr - N1) * DIMN + t * 4) = o;
}

// ---- 2) weight transpose fp32[K][N] -> bf16[N][K] --------------------------
__global__ void __launch_bounds__(256)
wtrans_kernel(const float* __restrict__ W, bf16_t* __restrict__ Wt, int K, int N) {
  const size_t idx = (size_t)blockIdx.x * 256 + threadIdx.x;
  if (idx >= (size_t)K * N) return;
  const int k = (int)(idx / N);
  const int n = (int)(idx - (size_t)k * N);
  Wt[(size_t)n * K + k] = (bf16_t)W[idx];
}

// ---- 3) bf16 WMMA GEMM: C[M][N] = A[M][K] * Bt[N][K]^T ---------------------
// 128x128 block tile, BK=32; 8 waves in 4(m) x 2(n) layout; each wave 2x4 tiles.
// Double-buffered LDS staged by async global->LDS DMA (4 b128 / thread / slab).
template <bool OUT_F32>
__global__ void __launch_bounds__(256)
gemm_kernel(const bf16_t* __restrict__ A, const bf16_t* __restrict__ Bt,
            void* __restrict__ Cv, int M, int N, int K, float scale) {
  const int m0 = blockIdx.x * 128;
  const int n0 = blockIdx.y * 128;
  const int t = threadIdx.x;
  const int w = t >> 5, lane = t & 31, hf = lane >> 4;
  const int wm = (w & 3) * 32, wn = (w >> 2) * 64;

  __shared__ __align__(16) bf16_t As[2][128 * 40];
  __shared__ __align__(16) bf16_t Bs[2][128 * 40];

  f32x8 acc[2][4] = {};

  const int r = t >> 1;                 // 0..127 : tile row
  const int hseg = (t & 1) * 16;        // two 16-elem halves of the 32-wide slab
  const bf16_t* ag = A  + (size_t)(m0 + r) * K + hseg;
  const bf16_t* bg = Bt + (size_t)(n0 + r) * K + hseg;
  const unsigned la[2] = { (unsigned)(uintptr_t)(As[0] + r * 40 + hseg),
                           (unsigned)(uintptr_t)(As[1] + r * 40 + hseg) };
  const unsigned lb[2] = { (unsigned)(uintptr_t)(Bs[0] + r * 40 + hseg),
                           (unsigned)(uintptr_t)(Bs[1] + r * 40 + hseg) };

  const int ksteps = K >> 5;
  // prologue: start DMA of slab 0 into buffer 0
  async_b128(la[0],      ag);
  async_b128(la[0] + 16, ag + 8);
  async_b128(lb[0],      bg);
  async_b128(lb[0] + 16, bg + 8);

  for (int kt = 0; kt < ksteps; ++kt) {
    if (kt + 1 < ksteps) {              // kick next slab into other buffer
      const int nb = (kt + 1) & 1;
      const bf16_t* ap = ag + (kt + 1) * 32;
      const bf16_t* bp = bg + (kt + 1) * 32;
      async_b128(la[nb],      ap);
      async_b128(la[nb] + 16, ap + 8);
      async_b128(lb[nb],      bp);
      async_b128(lb[nb] + 16, bp + 8);
      // async loads complete in order: <=4 outstanding => current slab landed
      asm volatile("s_wait_asynccnt 4" ::: "memory");
    } else {
      asm volatile("s_wait_asynccnt 0" ::: "memory");
    }
    __syncthreads();

    const bf16_t* Ab = As[kt & 1];
    const bf16_t* Bb = Bs[kt & 1];
    bf16x16 af[2], bfr[4];
    af[0] = frag_a(Ab + (wm) * 40, 40, lane);
    af[1] = frag_a(Ab + (wm + 16) * 40, 40, lane);
    #pragma unroll
    for (int j = 0; j < 4; ++j) bfr[j] = frag_b(Bb + (wn + j * 16) * 40, 40, lane);
    #pragma unroll
    for (int i = 0; i < 2; ++i)
      #pragma unroll
      for (int j = 0; j < 4; ++j)
        acc[i][j] = wmma_bf16(af[i], bfr[j], acc[i][j]);
    __syncthreads();                    // all waves done reading before next DMA reuse
  }

  // epilogue: C layout row = vg + 8*(lane>>4), col = lane&15 within each tile
  #pragma unroll
  for (int i = 0; i < 2; ++i)
    #pragma unroll
    for (int j = 0; j < 4; ++j) {
      const int row0 = m0 + wm + i * 16 + 8 * hf;
      const int col  = n0 + wn + j * 16 + (lane & 15);
      #pragma unroll
      for (int v = 0; v < 8; ++v) {
        float val = acc[i][j][v] * scale;
        if (OUT_F32) ((float*)Cv)[(size_t)(row0 + v) * N + col] = val;
        else         ((bf16_t*)Cv)[(size_t)(row0 + v) * N + col] = (bf16_t)val;
      }
    }
}

// ---- 4) flash attention per (b,h): Q 256x64 resident, stream K/V in 32-chunks
__global__ void __launch_bounds__(256)
attn_kernel(const bf16_t* __restrict__ Q, const bf16_t* __restrict__ KV,
            const int* __restrict__ mask, bf16_t* __restrict__ O) {
  const int b = blockIdx.x >> 4;
  const int h = blockIdx.x & 15;
  const int t = threadIdx.x;
  const int w = t >> 5, lane = t & 31, hf = lane >> 4;

  __shared__ __align__(16) bf16_t Qs[256 * 64];    // 32 KB
  __shared__ __align__(16) bf16_t Ks[32 * 72];     // K chunk  [j][d]
  __shared__ __align__(16) bf16_t Vt[64 * 40];     // V chunk transposed [d][j]
  __shared__ __align__(16) bf16_t Ps[8][16 * 40];  // per-wave P restage
  __shared__ __align__(16) bf16_t Vones[16 * 40];  // ones-tile: row n=0 all 1
  __shared__ int msk[32];

  { // load this head's Q (already *0.125 from q gemm)
    const bf16_t* qrow = Q + (size_t)(b * N2 + t) * (HEADS * DHEAD) + h * DHEAD;
    #pragma unroll
    for (int i = 0; i < 8; ++i)
      *(bf16x8*)(Qs + t * 64 + i * 8) = *(const bf16x8*)(qrow + i * 8);
  }
  if (t < 16) {
    #pragma unroll
    for (int k = 0; k < 32; ++k)
      Vones[t * 40 + k] = (t == 0) ? (bf16_t)1.0f : (bf16_t)0.0f;
  }
  __syncthreads();

  const bf16x16 onesb = frag_b(Vones, 40, lane);

  // acc[qt][0..3] = output tiles; acc[qt][4] = running row-sum (col 0)
  f32x8 acc[2][5] = {};
  float rmax[2][8];
  #pragma unroll
  for (int qt = 0; qt < 2; ++qt)
    #pragma unroll
    for (int v = 0; v < 8; ++v) rmax[qt][v] = -1e30f;

  for (int j0 = 0; j0 < NTOT; j0 += 32) {
    // cooperative stage of K (row-major) and V (transposed) chunks
    const int jr = t >> 3, seg = t & 7;
    const size_t rowbase = (size_t)(b * NTOT + j0 + jr) * (2 * HEADS * DHEAD);
    bf16x8 k8 = *(const bf16x8*)(KV + rowbase + h * DHEAD + seg * 8);
    *(bf16x8*)(Ks + jr * 72 + seg * 8) = k8;
    bf16x8 v8 = *(const bf16x8*)(KV + rowbase + HEADS * DHEAD + h * DHEAD + seg * 8);
    #pragma unroll
    for (int e = 0; e < 8; ++e) Vt[(seg * 8 + e) * 40 + jr] = v8[e];
    if (t < 32) msk[t] = mask[(size_t)b * NTOT + j0 + t];
    if (j0 + 32 < NTOT) {   // global_prefetch_b8 next chunk
      __builtin_prefetch(KV + rowbase + 32 * (2 * HEADS * DHEAD) + h * DHEAD, 0, 0);
      __builtin_prefetch(KV + rowbase + 32 * (2 * HEADS * DHEAD) + HEADS * DHEAD + h * DHEAD, 0, 0);
    }
    __syncthreads();

    bf16x16 kb00 = frag_b(Ks + 0 * 72 + 0,  72, lane);   // j 0-15,  d 0-31
    bf16x16 kb01 = frag_b(Ks + 0 * 72 + 32, 72, lane);   // j 0-15,  d 32-63
    bf16x16 kb10 = frag_b(Ks + 16 * 72 + 0, 72, lane);   // j 16-31, d 0-31
    bf16x16 kb11 = frag_b(Ks + 16 * 72 + 32, 72, lane);  // j 16-31, d 32-63
    bf16x16 vb[4];
    #pragma unroll
    for (int d = 0; d < 4; ++d) vb[d] = frag_b(Vt + (d * 16) * 40, 40, lane);
    const int mk0 = msk[lane & 15];
    const int mk1 = msk[16 + (lane & 15)];

    #pragma unroll
    for (int qt = 0; qt < 2; ++qt) {
      const int qbase = w * 32 + qt * 16;
      bf16x16 qa0 = frag_a(Qs + qbase * 64, 64, lane);
      bf16x16 qa1 = frag_a(Qs + qbase * 64 + 32, 64, lane);
      f32x8 s0 = {}; s0 = wmma_bf16(qa0, kb00, s0); s0 = wmma_bf16(qa1, kb01, s0);
      f32x8 s1 = {}; s1 = wmma_bf16(qa0, kb10, s1); s1 = wmma_bf16(qa1, kb11, s1);

      #pragma unroll
      for (int v = 0; v < 8; ++v) {
        float a0 = (mk0 == 0) ? -10000.f : s0[v];
        float a1 = (mk1 == 0) ? -10000.f : s1[v];
        // row max across the 16 lanes of this half (rows = v + 8*hf)
        float r = fmaxf(a0, a1);
        r = fmaxf(r, __shfl_xor(r, 1));
        r = fmaxf(r, __shfl_xor(r, 2));
        r = fmaxf(r, __shfl_xor(r, 4));
        r = fmaxf(r, __shfl_xor(r, 8));
        const float nm   = fmaxf(rmax[qt][v], r);
        const float corr = __expf(rmax[qt][v] - nm);
        rmax[qt][v] = nm;
        const float p0 = __expf(a0 - nm);
        const float p1 = __expf(a1 - nm);
        const int prow = v + 8 * hf;
        Ps[w][prow * 40 + (lane & 15)]      = (bf16_t)p0;
        Ps[w][prow * 40 + 16 + (lane & 15)] = (bf16_t)p1;
        acc[qt][0][v] *= corr;
        acc[qt][1][v] *= corr;
        acc[qt][2][v] *= corr;
        acc[qt][3][v] *= corr;
        acc[qt][4][v] *= corr;   // running denominator rescales identically
      }
      // same-wave LDS store -> load ordering
      asm volatile("s_wait_dscnt 0" ::: "memory");
      bf16x16 pa = frag_a(Ps[w], 40, lane);
      #pragma unroll
      for (int d = 0; d < 4; ++d) acc[qt][d] = wmma_bf16(pa, vb[d], acc[qt][d]);
      acc[qt][4] = wmma_bf16(pa, onesb, acc[qt][4]);  // row-sum on the matrix pipe
    }
    __syncthreads();
  }

  // normalize (denominator sits in column 0 of acc[qt][4]) and write out
  #pragma unroll
  for (int qt = 0; qt < 2; ++qt) {
    float rinv[8];
    #pragma unroll
    for (int v = 0; v < 8; ++v)
      rinv[v] = 1.f / __shfl(acc[qt][4][v], hf << 4);   // bcast col 0 of this half
    #pragma unroll
    for (int d = 0; d < 4; ++d)
      #pragma unroll
      for (int v = 0; v < 8; ++v) {
        const int row = w * 32 + qt * 16 + v + 8 * hf;
        const int col = d * 16 + (lane & 15);
        const float val = acc[qt][d][v] * rinv[v];
        O[(size_t)(b * N2 + row) * (HEADS * DHEAD) + h * DHEAD + col] = (bf16_t)val;
      }
  }
}

// ---------------------------------------------------------------------------
extern "C" void kernel_launch(void* const* d_in, const int* in_sizes, int n_in,
                              void* d_out, int out_size, void* d_ws, size_t ws_size,
                              hipStream_t stream) {
  (void)in_sizes; (void)n_in; (void)out_size; (void)ws_size;
  const float* x    = (const float*)d_in[0];
  const float* lat  = (const float*)d_in[1];
  const int*   mask = (const int*)d_in[2];
  const float* gm   = (const float*)d_in[3];
  const float* bm   = (const float*)d_in[4];
  const float* gl   = (const float*)d_in[5];
  const float* bl   = (const float*)d_in[6];
  const float* Wq   = (const float*)d_in[7];
  const float* Wkv  = (const float*)d_in[8];
  const float* Wout = (const float*)d_in[9];
  float* out = (float*)d_out;

  char* ws = (char*)d_ws;
  auto carve = [&](size_t bytes) {
    char* p = ws;
    ws += (bytes + 255) & ~(size_t)255;
    return p;
  };
  bf16_t* kvin  = (bf16_t*)carve((size_t)BATCH * NTOT * DIMN * 2);      // 138 MB
  bf16_t* lnbuf = (bf16_t*)carve((size_t)BATCH * N2 * DIMN * 2);        //   4 MB
  bf16_t* WqT   = (bf16_t*)carve((size_t)DIMN * DIMN * 2);              //   2 MB
  bf16_t* WkvT  = (bf16_t*)carve((size_t)DIMN * 2 * DIMN * 2);          //   4 MB
  bf16_t* WoutT = (bf16_t*)carve((size_t)DIMN * DIMN * 2);              //   2 MB
  bf16_t* Qb    = (bf16_t*)carve((size_t)BATCH * N2 * DIMN * 2);        //   4 MB
  bf16_t* KVb   = (bf16_t*)carve((size_t)BATCH * NTOT * 2 * DIMN * 2);  // 277 MB
  bf16_t* Ob    = (bf16_t*)carve((size_t)BATCH * N2 * DIMN * 2);        //   4 MB

  ln_kernel<<<BATCH * NTOT, 256, 0, stream>>>(x, lat, gm, bm, gl, bl, kvin, lnbuf);
  wtrans_kernel<<<(DIMN * DIMN + 255) / 256, 256, 0, stream>>>(Wq, WqT, DIMN, DIMN);
  wtrans_kernel<<<(DIMN * 2 * DIMN + 255) / 256, 256, 0, stream>>>(Wkv, WkvT, DIMN, 2 * DIMN);
  wtrans_kernel<<<(DIMN * DIMN + 255) / 256, 256, 0, stream>>>(Wout, WoutT, DIMN, DIMN);

  // q = ln @ Wq, scale 64^-0.5 folded in; M=2048 N=1024
  gemm_kernel<false><<<dim3((BATCH * N2) / 128, DIMN / 128), 256, 0, stream>>>(
      lnbuf, WqT, Qb, BATCH * N2, DIMN, DIMN, 0.125f);
  // kv = [xn; ln] @ Wkv; M=67584 N=2048 (dominant GEMM, 283 GFLOP)
  gemm_kernel<false><<<dim3((BATCH * NTOT) / 128, (2 * DIMN) / 128), 256, 0, stream>>>(
      kvin, WkvT, KVb, BATCH * NTOT, 2 * DIMN, DIMN, 1.0f);

  attn_kernel<<<BATCH * HEADS, 256, 0, stream>>>(Qb, KVb, mask, Ob);

  // final projection in fp32
  gemm_kernel<true><<<dim3((BATCH * N2) / 128, DIMN / 128), 256, 0, stream>>>(
      Ob, WoutT, out, BATCH * N2, DIMN, DIMN, 1.0f);
}
